// CorrCMFD_60653528154739
// MI455X (gfx1250) — matrix operand order repaired and checked by reading, not compiled, not run on hardware
//
#include <hip/hip_runtime.h>
#include <hip/hip_bf16.h>
#include <math.h>

typedef __attribute__((ext_vector_type(16))) __bf16 v16bf;
typedef __attribute__((ext_vector_type(8)))  __bf16 v8bf;
typedef __attribute__((ext_vector_type(8)))  float  v8f;
typedef __attribute__((ext_vector_type(4)))  unsigned int v4u;
typedef __attribute__((ext_vector_type(8)))  int v8i;
typedef __attribute__((ext_vector_type(4)))  int v4i;

#define B_    4
#define C_    256
#define HW_   1024          // 32*32
#define SDIM  32

#if defined(__gfx1250__) && __has_builtin(__builtin_amdgcn_tensor_load_to_lds) && \
    __has_builtin(__builtin_amdgcn_s_wait_tensorcnt)
#define USE_TDM 1
#else
#define USE_TDM 0
#endif

// LDS row stride for staged tiles: 32 bf16 of data + 8 bf16 pad = 40 (80 B).
// 80 B row stride keeps 16 B alignment for b128 reads and rotates banks
// (20 DWORDs -> 16 distinct banks over 16 consecutive rows).
#define KPAD 40

// ---------------------------------------------------------------------------
// Kernel 1: channel-L2 normalize x; write bf16 Xt in (b, n, c) layout so both
// GEMM operands are row-major 2-D tiles with K contiguous (TDM-friendly).
// ---------------------------------------------------------------------------
__global__ void norm_pack_kernel(const float* __restrict__ x,
                                 __bf16* __restrict__ xt) {
  int t = blockIdx.x * blockDim.x + threadIdx.x;   // 0..4095
  int b = t >> 10;
  int n = t & (HW_ - 1);
  const float* xb = x + (size_t)b * C_ * HW_ + n;  // lanes stride n: coalesced
  float ss = 0.0f;
#pragma unroll 4
  for (int c = 0; c < C_; ++c) {
    float v = xb[(size_t)c * HW_];
    ss += v * v;
  }
  float inv = 1.0f / fmaxf(sqrtf(ss), 1e-12f);
  __bf16* ot = xt + ((size_t)b * HW_ + n) * C_;    // row n, contiguous in c
#pragma unroll 4
  for (int c = 0; c < C_; ++c)
    ot[c] = (__bf16)(xb[(size_t)c * HW_] * inv);
}

// ---------------------------------------------------------------------------
// TDM: DMA a (rows x 32) bf16 sub-tile of Xt (row stride 256 elems) into LDS
// with 4-DWORD padding after every 16 DWORDs (=> KPAD row stride in LDS).
// D# fields per ISA 8.3/8.4. Issued by one wave; EXEC-independent.
// ---------------------------------------------------------------------------
#if USE_TDM
__device__ __forceinline__ unsigned lds_off(const void* p) {
  return (unsigned)(size_t)(__attribute__((address_space(3))) const char*)p;
}

__device__ __forceinline__ void tdm_load_tile(unsigned lds_addr,
                                              const void* gsrc, int rows) {
  unsigned long long ga = (unsigned long long)gsrc;
  v4u g0;
  g0[0] = 1u;                                   // count=1 (valid user D#)
  g0[1] = lds_addr;                             // LDS byte address
  g0[2] = (unsigned)ga;                         // global_addr[31:0]
  g0[3] = (unsigned)((ga >> 32) & 0x01FFFFFFull) | (2u << 30);  // [56:32]|type=2
  v8i g1;
  g1[0] = (int)0x06D10000;   // data_size=2B | pad_en | pad_interval=16dw | pad_amount=4dw
  g1[1] = (int)(256u << 16); // tensor_dim0 = 256 (K extent)        bits[79:48]
  g1[2] = (int)(1024u << 16);// tensor_dim1 = 1024 (rows extent)    bits[111:80]
  g1[3] = (int)(32u << 16);  // tile_dim0 = 32 K elements           bits[127:112]
  g1[4] = rows;              // tile_dim1 = rows                    bits[143:128]
  g1[5] = 256;               // tensor_dim0_stride = 256 elems      bits[207:160]
  g1[6] = 0;
  g1[7] = 0;
  // 6-arg toolchain variant: (g0, g1, g2, g3, g4, cpol)
  __builtin_amdgcn_tensor_load_to_lds(g0, g1, (v4i)0, (v4i)0, (v8i)0, 0);
}
#endif

// ---------------------------------------------------------------------------
// Kernel 2: aff = Xn * Xn^T per batch (1024x1024x256) via
// v_wmma_f32_16x16x32_bf16 with the zero_window mask fused in the epilogue.
// Block = 256 threads = 8 wave32; macro-tile 32(M) x 64(N); K stepped by 32.
// Tiles staged into LDS by the Tensor Data Mover (fallback: u32 copies).
// ---------------------------------------------------------------------------
__global__ void affinity_wmma_kernel(const __bf16* __restrict__ xt,
                                     float* __restrict__ aff) {
  __shared__ __bf16 As[32][KPAD];   // [m][k]
  __shared__ __bf16 Bs[64][KPAD];   // [n][k]

  const int b   = blockIdx.z;
  const int bm  = blockIdx.y;     // 0..31 -> M tile of 32
  const int bn  = blockIdx.x;     // 0..15 -> N tile of 64
  const int tid = threadIdx.x;
  const int lane  = tid & 31;
  const int wv    = tid >> 5;     // 0..7
  const int wm    = wv & 1;       // 2 waves along M
  const int wn    = wv >> 1;      // 4 waves along N
  const int lhalf = lane >> 4;    // 0/1
  const int lrow  = lane & 15;

  const __bf16* Xt = xt + (size_t)b * HW_ * C_;
  v8f acc = {};

  for (int kt = 0; kt < C_ / 32; ++kt) {
    __syncthreads();              // previous tile fully consumed
#if USE_TDM
    if (wv == 0) {
      tdm_load_tile(lds_off(&As[0][0]),
                    Xt + (size_t)(bm * 32) * C_ + kt * 32, 32);
      tdm_load_tile(lds_off(&Bs[0][0]),
                    Xt + (size_t)(bn * 64) * C_ + kt * 32, 64);
      __builtin_amdgcn_s_wait_tensorcnt(0);
    }
#else
    // Fallback: u32 (2xbf16) copies into the same padded layout.
    for (int idx = tid; idx < 32 * 16; idx += 256) {
      int m = idx >> 4, kp = idx & 15;
      *(unsigned*)&As[m][kp * 2] =
          *(const unsigned*)(Xt + (size_t)(bm * 32 + m) * C_ + kt * 32 + kp * 2);
    }
    for (int idx = tid; idx < 64 * 16; idx += 256) {
      int n = idx >> 4, kp = idx & 15;
      *(unsigned*)&Bs[n][kp * 2] =
          *(const unsigned*)(Xt + (size_t)(bn * 64 + n) * C_ + kt * 32 + kp * 2);
    }
#endif
    __syncthreads();

    // Fragments per ISA 7.12.2; contiguous 16 B runs -> ds_load_b128.
    // A (16-bit 16x32): lane half selects K sub-block {0..7 / 8..15} and
    // {16..23 / 24..31}.
    v8bf alo = *(const v8bf*)&As[wm * 16 + lrow][lhalf * 8];
    v8bf ahi = *(const v8bf*)&As[wm * 16 + lrow][16 + lhalf * 8];
    v16bf afrag = __builtin_shufflevector(alo, ahi, 0, 1, 2, 3, 4, 5, 6, 7, 8,
                                          9, 10, 11, 12, 13, 14, 15);
    // B (16-bit 32x16): same K sub-block order as A.
    v8bf blo = *(const v8bf*)&Bs[wn * 16 + lrow][lhalf * 8];
    v8bf bhi = *(const v8bf*)&Bs[wn * 16 + lrow][16 + lhalf * 8];
    v16bf bfrag = __builtin_shufflevector(blo, bhi, 0, 1, 2, 3, 4, 5, 6, 7, 8,
                                          9, 10, 11, 12, 13, 14, 15);

    acc = __builtin_amdgcn_wmma_f32_16x16x32_bf16(
        /*neg_a=*/false, afrag, /*neg_b=*/false, bfrag,
        /*c_mod=*/(short)0, acc, /*reuse_a=*/false, /*reuse_b=*/false);
  }

  // Epilogue: fused zero_window mask. sr = sc = 32*0.05 = 1.6.
  const float inv2s2 = 1.0f / (2.0f * 1.6f * 1.6f);
#pragma unroll
  for (int v = 0; v < 8; ++v) {
    int Mg = bm * 32 + wm * 16 + v + 8 * lhalf;   // query index n (row)
    int Ng = bn * 64 + wn * 16 + lrow;            // key index m (col)
    float cr = (float)(Mg >> 5), cc = (float)(Mg & 31);
    float rr = (float)(Ng >> 5), rc = (float)(Ng & 31);
    float g = 1.0f - __expf(-(rr - cr) * (rr - cr) * inv2s2) *
                     __expf(-(rc - cc) * (rc - cc) * inv2s2);
    aff[(((size_t)b * HW_ + Mg) << 10) + Ng] = acc[v] * g;
  }
}

// ---------------------------------------------------------------------------
// Kernel 3: fused conv4d(conv4d(aff, w1, b1), w2, b2) -> y, all through LDS.
// Output tile (l,d,h,w) = (4,4,8,8). LDS: aff halo 8*8*12*12 f32 (36.9 KB) +
// y1 halo 6*6*10*10*16 f32 (230.4 KB) + weights (10.6 KB) = ~278 KB
// -> needs CDNA5's 320 KB WGP LDS (dynamic shared).
// Bias semantics: per-frame validity over the l kernel dim only.
// ---------------------------------------------------------------------------
#define AL 8
#define AD 8
#define AH 12
#define AW 12
#define YL 6
#define YD 6
#define YH 10
#define YW 10
#define NAFF (AL*AD*AH*AW)          // 9216
#define NY1  (YL*YD*YH*YW*16)       // 57600

__global__ void fused_conv4d_kernel(const float* __restrict__ aff,
                                    const float* __restrict__ w1,
                                    const float* __restrict__ b1,
                                    const float* __restrict__ w2,
                                    const float* __restrict__ b2,
                                    float* __restrict__ y) {
  extern __shared__ float smem[];
  float* s_aff = smem;                // NAFF
  float* s_y1  = s_aff + NAFF;        // NY1
  float* s_w1  = s_y1 + NY1;          // 1296
  float* s_b1  = s_w1 + 1296;         // 48
  float* s_w2  = s_b1 + 48;           // 1296
  float* s_b2  = s_w2 + 1296;         // 3

  const int tid = threadIdx.x;
  const int b   = blockIdx.z;
  const int l0  = (blockIdx.y >> 3) * 4;
  const int d0  = (blockIdx.y & 7) * 4;
  const int h0  = (blockIdx.x >> 2) * 8;
  const int w0  = (blockIdx.x & 3) * 8;

  for (int i = tid; i < 1296; i += 256) { s_w1[i] = w1[i]; s_w2[i] = w2[i]; }
  for (int i = tid; i < 48; i += 256) s_b1[i] = b1[i];
  if (tid < 3) s_b2[tid] = b2[tid];

  // Stage aff halo (zero-fill = conv1 spatial zero padding).
  const float* affb = aff + ((size_t)b * HW_ << 10);
  for (int idx = tid; idx < NAFF; idx += 256) {
    int aw = idx % AW, ah = (idx / AW) % AH, ad = (idx / (AW * AH)) % AD,
        al = idx / (AW * AH * AD);
    int gl = l0 + al - 2, gd = d0 + ad - 2, gh = h0 + ah - 2, gw = w0 + aw - 2;
    float v = 0.0f;
    if (gl >= 0 && gl < SDIM && gd >= 0 && gd < SDIM &&
        gh >= 0 && gh < SDIM && gw >= 0 && gw < SDIM)
      v = affb[(((size_t)gl * SDIM + gd) << 10) + gh * SDIM + gw];
    s_aff[idx] = v;
  }
  __syncthreads();

  // conv1: 16-channel y1 halo tile (zero outside volume = conv2 zero padding).
  for (int idx = tid; idx < NY1; idx += 256) {
    int ch = idx & 15, r = idx >> 4;
    int yw = r % YW, yh = (r / YW) % YH, yd = (r / (YW * YH)) % YD,
        yl = r / (YW * YH * YD);
    int gl = l0 + yl - 1, gd = d0 + yd - 1, gh = h0 + yh - 1, gw = w0 + yw - 1;
    float s = 0.0f;
    if (gl >= 0 && gl < SDIM && gd >= 0 && gd < SDIM &&
        gh >= 0 && gh < SDIM && gw >= 0 && gw < SDIM) {
#pragma unroll
      for (int i = 0; i < 3; ++i) {          // bias: once per VALID l-tap
        int j = gl + i - 1;
        if (j >= 0 && j < SDIM) s += s_b1[i * 16 + ch];
      }
      const float* wch = s_w1 + ch * 81;
#pragma unroll
      for (int i = 0; i < 3; ++i)
#pragma unroll
        for (int j = 0; j < 3; ++j)
#pragma unroll
          for (int p = 0; p < 3; ++p)
#pragma unroll
            for (int q = 0; q < 3; ++q)
              s = fmaf(s_aff[(yl + i) * (AD * AH * AW) + (yd + j) * (AH * AW) +
                             (yh + p) * AW + (yw + q)],
                       wch[i * 27 + j * 9 + p * 3 + q], s);
    }
    s_y1[idx] = s;
  }
  __syncthreads();

  // conv2: 16ch * 3^4 reduce -> 4 outputs per thread.
  for (int idx = tid; idx < 1024; idx += 256) {
    int ow = idx & 7, oh = (idx >> 3) & 7, od = (idx >> 6) & 3, ol = idx >> 8;
    int gl = l0 + ol, gd = d0 + od, gh = h0 + oh, gw = w0 + ow;
    float acc = 0.0f;
#pragma unroll
    for (int i = 0; i < 3; ++i) {
      int j = gl + i - 1;
      if (j >= 0 && j < SDIM) acc += s_b2[i];
    }
    for (int ch = 0; ch < 16; ++ch) {
      const float* wch = s_w2 + ch * 81;
#pragma unroll
      for (int i = 0; i < 3; ++i)
#pragma unroll
        for (int j = 0; j < 3; ++j)
#pragma unroll
          for (int p = 0; p < 3; ++p)
#pragma unroll
            for (int q = 0; q < 3; ++q)
              acc = fmaf(
                  s_y1[((((ol + i) * YD + (od + j)) * YH + (oh + p)) * YW +
                        (ow + q)) * 16 + ch],
                  wch[i * 27 + j * 9 + p * 3 + q], acc);
    }
    y[(((size_t)b * HW_ + gl * SDIM + gd) << 10) + gh * SDIM + gw] = acc;
  }
}

// ---------------------------------------------------------------------------
// Kernel 4: top-3 over the 1024 query dim per (b, pixel). One wave32 per
// pixel; per-lane top-3 then shuffle-tree merge (warpSize == 32 on gfx1250).
// ---------------------------------------------------------------------------
__device__ __forceinline__ void ins3(float v, float& a, float& b, float& c) {
  if (v > a)      { c = b; b = a; a = v; }
  else if (v > b) { c = b; b = v; }
  else if (v > c) { c = v; }
}

__global__ void topk_kernel(const float* __restrict__ y,
                            float* __restrict__ out) {
  int b = blockIdx.x >> 10;
  int p = blockIdx.x & (HW_ - 1);
  int lane = threadIdx.x;
  float v0 = -INFINITY, v1 = -INFINITY, v2 = -INFINITY;
  const float* yb = y + ((size_t)b * HW_ << 10) + p;
  for (int n = lane; n < HW_; n += 32)
    ins3(yb[(size_t)n << 10], v0, v1, v2);
#pragma unroll
  for (int off = 16; off > 0; off >>= 1) {
    float o0 = __shfl_xor(v0, off);
    float o1 = __shfl_xor(v1, off);
    float o2 = __shfl_xor(v2, off);
    ins3(o0, v0, v1, v2); ins3(o1, v0, v1, v2); ins3(o2, v0, v1, v2);
  }
  if (lane == 0) {
    out[((size_t)(b * 3 + 0) << 10) + p] = v0;
    out[((size_t)(b * 3 + 1) << 10) + p] = v1;
    out[((size_t)(b * 3 + 2) << 10) + p] = v2;
  }
}

// ---------------------------------------------------------------------------
extern "C" void kernel_launch(void* const* d_in, const int* in_sizes, int n_in,
                              void* d_out, int out_size, void* d_ws, size_t ws_size,
                              hipStream_t stream) {
  const float* x  = (const float*)d_in[0];
  const float* w1 = (const float*)d_in[1];
  const float* b1 = (const float*)d_in[2];
  const float* w2 = (const float*)d_in[3];
  const float* b2 = (const float*)d_in[4];
  float* out = (float*)d_out;

  char* ws = (char*)d_ws;
  __bf16* xt  = (__bf16*)ws;                               // 4*1024*256*2 = 2 MB
  float*  aff = (float*)(ws + (size_t)2 * 1024 * 1024);    // 16 MB
  float*  y   = (float*)(ws + (size_t)18 * 1024 * 1024);   // 16 MB

  norm_pack_kernel<<<16, 256, 0, stream>>>(x, xt);

  affinity_wmma_kernel<<<dim3(16, 32, B_), 256, 0, stream>>>(xt, aff);

  size_t smem = (size_t)(NAFF + NY1 + 1296 + 48 + 1296 + 3) * sizeof(float);
  fused_conv4d_kernel<<<dim3(16, 64, B_), 256, smem, stream>>>(aff, w1, b1, w2,
                                                               b2, y);

  topk_kernel<<<B_ * HW_, 32, 0, stream>>>(y, out);
}
